// HeteroHGNN_11055245820286
// MI455X (gfx1250) — compile-verified, matrix operand order checked by reading, not run on hardware
//
#include <hip/hip_runtime.h>

#define NN 50000
#define DD 128
#define EE 800000
#define NEG_SLOPE 0.2f

typedef __attribute__((ext_vector_type(16))) __bf16 v16bf;
typedef __attribute__((ext_vector_type(8)))  __bf16 v8bf;
typedef __attribute__((ext_vector_type(8)))  float  v8f;

// ---------- helpers ----------

__device__ inline __bf16 f2bf(float f) {
  union { float f; unsigned u; } in; in.f = f;
  unsigned u = in.u;
  unsigned r = (u + 0x7fffu + ((u >> 16) & 1u)) >> 16;  // round-to-nearest-even
  union { unsigned short s; __bf16 b; } out; out.s = (unsigned short)r;
  return out.b;
}

// monotone float <-> int encoding (involution) so integer atomicMax == float max
__device__ inline int enc_f32(float f) {
  int i = __float_as_int(f);
  return (i >= 0) ? i : (i ^ 0x7fffffff);
}
__device__ inline float dec_f32(int k) {
  int i = (k >= 0) ? k : (k ^ 0x7fffffff);
  return __int_as_float(i);
}

__device__ inline float leaky(float e) { return e > 0.0f ? e : NEG_SLOPE * e; }

// Load one 16x32 bf16 fragment: per-lane K is {0..7} and {16..23} relative to
// base (lanes>=16 already shifted by +8 by caller). Two 16B LDS loads.
__device__ inline v16bf load_frag16(const __bf16* base) {
  v8bf lo = *(const v8bf*)(base);
  v8bf hi = *(const v8bf*)(base + 16);
  return __builtin_shufflevector(lo, hi, 0,1,2,3,4,5,6,7,8,9,10,11,12,13,14,15);
}

// ---------- WMMA GEMM: out[r,c] (+)= sum_k A[r,k] * W[k,c], A:[nrows x 128], W:[128 x 128] ----------

template <bool ACCUM>
__global__ __launch_bounds__(128) void wmma_gemm_kernel(
    const float* __restrict__ A, const float* __restrict__ W,
    float* __restrict__ out, int nrows)
{
  __shared__ __align__(16) __bf16 sA[64 * 128];    // row-major: sA[row*128 + k]
  __shared__ __align__(16) __bf16 sW[128 * 128];   // transposed: sW[col*128 + k]

  const int tid  = threadIdx.x;
  const int row0 = blockIdx.x * 64;

  // stage W (transposed) as bf16
  for (int i = tid; i < 128 * 128; i += 128) {
    int k = i >> 7, c = i & 127;
    sW[c * 128 + k] = f2bf(W[i]);
  }
  // stage X tile as bf16 (zero-pad rows past nrows)
  for (int i = tid; i < 64 * 128; i += 128) {
    int r = i >> 7;
    int gr = row0 + r;
    float v = (gr < nrows) ? A[(size_t)gr * 128 + (i & 127)] : 0.0f;
    sA[i] = f2bf(v);
  }
  __syncthreads();

  const int wave  = tid >> 5;
  const int lane  = tid & 31;
  const int lmod  = lane & 15;
  const int lhalf = (lane >= 16) ? 8 : 0;   // K sub-offset per documented layout
  const int m0    = wave * 16;
  const bool full = (row0 + 64 <= nrows);   // uniform: all but the last block

  const __bf16* aBase = &sA[(m0 + lmod) * 128];

  // A fragments for all four K-steps stay resident across the N loop
  v16bf a0 = load_frag16(aBase +  0 + lhalf);
  v16bf a1 = load_frag16(aBase + 32 + lhalf);
  v16bf a2 = load_frag16(aBase + 64 + lhalf);
  v16bf a3 = load_frag16(aBase + 96 + lhalf);

  for (int n0 = 0; n0 < 128; n0 += 16) {
    const __bf16* bBase = &sW[(n0 + lmod) * 128];
    v8f c = {};
    c = __builtin_amdgcn_wmma_f32_16x16x32_bf16(false, a0, false,
        load_frag16(bBase +  0 + lhalf), (short)0, c, false, false);
    c = __builtin_amdgcn_wmma_f32_16x16x32_bf16(false, a1, false,
        load_frag16(bBase + 32 + lhalf), (short)0, c, false, false);
    c = __builtin_amdgcn_wmma_f32_16x16x32_bf16(false, a2, false,
        load_frag16(bBase + 64 + lhalf), (short)0, c, false, false);
    c = __builtin_amdgcn_wmma_f32_16x16x32_bf16(false, a3, false,
        load_frag16(bBase + 96 + lhalf), (short)0, c, false, false);

    // D layout: lanes 0-15 -> M = r, lanes 16-31 -> M = 8 + r ; N = lane%16
    const int col   = n0 + lmod;
    const int rbase = row0 + m0 + ((lane >= 16) ? 8 : 0);
    float* o = out + (size_t)rbase * 128 + col;
    if (full) {
#pragma unroll
      for (int r = 0; r < 8; ++r) {
        if (ACCUM) o[(size_t)r * 128] += c[r];
        else       o[(size_t)r * 128]  = c[r];
      }
    } else {
#pragma unroll
      for (int r = 0; r < 8; ++r) {
        if (rbase + r < nrows) {
          if (ACCUM) o[(size_t)r * 128] += c[r];
          else       o[(size_t)r * 128]  = c[r];
        }
      }
    }
  }
}

// ---------- fills ----------

__global__ void fill_f32_kernel(float* p, int n, float v) {
  int i = blockIdx.x * blockDim.x + threadIdx.x;
  if (i < n) p[i] = v;
}
__global__ void fill_i32_kernel(int* p, int n, int v) {
  int i = blockIdx.x * blockDim.x + threadIdx.x;
  if (i < n) p[i] = v;
}

// ---------- GAT attention scalars: as[n] = h[n].a_src, ad[n] = h[n].a_dst ----------

__global__ void alpha_dots_kernel(const float* __restrict__ h,
                                  const float* __restrict__ a_src,
                                  const float* __restrict__ a_dst,
                                  float* __restrict__ as_, float* __restrict__ ad_,
                                  int n)
{
  int gw   = (blockIdx.x * blockDim.x + threadIdx.x) >> 5;
  int lane = threadIdx.x & 31;
  if (gw >= n) return;
  float4 hv = ((const float4*)(h + (size_t)gw * 128))[lane];
  float4 s4 = ((const float4*)a_src)[lane];
  float4 d4 = ((const float4*)a_dst)[lane];
  float ss = hv.x * s4.x + hv.y * s4.y + hv.z * s4.z + hv.w * s4.w;
  float dd = hv.x * d4.x + hv.y * d4.y + hv.z * d4.z + hv.w * d4.w;
#pragma unroll
  for (int m = 16; m >= 1; m >>= 1) {
    ss += __shfl_xor(ss, m, 32);
    dd += __shfl_xor(dd, m, 32);
  }
  if (lane == 0) { as_[gw] = ss; ad_[gw] = dd; }
}

// ---------- edge softmax passes (edges = E real + N self-loops) ----------

__global__ void edge_max_kernel(const int* __restrict__ src, const int* __restrict__ dst,
                                const float* __restrict__ as_, const float* __restrict__ ad_,
                                int* __restrict__ m_enc, int e_cnt, int n_cnt)
{
  int i = blockIdx.x * blockDim.x + threadIdx.x;
  if (i >= e_cnt + n_cnt) return;
  int s = (i < e_cnt) ? src[i] : (i - e_cnt);
  int d = (i < e_cnt) ? dst[i] : (i - e_cnt);
  float e = leaky(as_[s] + ad_[d]);
  atomicMax(&m_enc[d], enc_f32(e));
}

__global__ void edge_expsum_kernel(const int* __restrict__ src, const int* __restrict__ dst,
                                   const float* __restrict__ as_, const float* __restrict__ ad_,
                                   const int* __restrict__ m_enc, float* __restrict__ denom,
                                   int e_cnt, int n_cnt)
{
  int i = blockIdx.x * blockDim.x + threadIdx.x;
  if (i >= e_cnt + n_cnt) return;
  int s = (i < e_cnt) ? src[i] : (i - e_cnt);
  int d = (i < e_cnt) ? dst[i] : (i - e_cnt);
  float e = leaky(as_[s] + ad_[d]);
  atomicAdd(&denom[d], expf(e - dec_f32(m_enc[d])));
}

// one wave per edge: acc[d,:] += alpha * h[s,:]  (4 coalesced f32 atomics / lane)
__global__ void edge_scatter_kernel(const int* __restrict__ src, const int* __restrict__ dst,
                                    const float* __restrict__ as_, const float* __restrict__ ad_,
                                    const int* __restrict__ m_enc, const float* __restrict__ denom,
                                    const float* __restrict__ h, float* __restrict__ acc,
                                    int e_cnt, int n_cnt)
{
  int gw   = (blockIdx.x * blockDim.x + threadIdx.x) >> 5;
  int lane = threadIdx.x & 31;
  if (gw >= e_cnt + n_cnt) return;
  int s = (gw < e_cnt) ? src[gw] : (gw - e_cnt);
  int d = (gw < e_cnt) ? dst[gw] : (gw - e_cnt);
  float e     = leaky(as_[s] + ad_[d]);
  float alpha = expf(e - dec_f32(m_enc[d])) / denom[d];
  float4 v = ((const float4*)(h + (size_t)s * 128))[lane];
  float* o = acc + (size_t)d * 128 + lane * 4;
  atomicAdd(o + 0, alpha * v.x);
  atomicAdd(o + 1, alpha * v.y);
  atomicAdd(o + 2, alpha * v.z);
  atomicAdd(o + 3, alpha * v.w);
}

// ---------- SAGE aggregation ----------

__global__ void sage_sum_kernel(const int* __restrict__ src, const int* __restrict__ dst,
                                const float* __restrict__ x, float* __restrict__ ssum,
                                float* __restrict__ cnt, int e_cnt)
{
  int gw   = (blockIdx.x * blockDim.x + threadIdx.x) >> 5;
  int lane = threadIdx.x & 31;
  if (gw >= e_cnt) return;
  int s = src[gw], d = dst[gw];
  float4 v = ((const float4*)(x + (size_t)s * 128))[lane];
  float* o = ssum + (size_t)d * 128 + lane * 4;
  atomicAdd(o + 0, v.x);
  atomicAdd(o + 1, v.y);
  atomicAdd(o + 2, v.z);
  atomicAdd(o + 3, v.w);
  if (lane == 0) atomicAdd(&cnt[d], 1.0f);
}

__global__ void sage_mean_kernel(float* __restrict__ ssum, const float* __restrict__ cnt,
                                 int total)
{
  int i = blockIdx.x * blockDim.x + threadIdx.x;
  if (i >= total) return;
  ssum[i] = ssum[i] / fmaxf(cnt[i >> 7], 1.0f);
}

// ---------- final: out = (acc + b1 + b2 + b3) / 3 ----------

__global__ void finalize_kernel(const float* __restrict__ acc,
                                const float* __restrict__ b1, const float* __restrict__ b2,
                                const float* __restrict__ b3, float* __restrict__ out,
                                int total)
{
  int i = blockIdx.x * blockDim.x + threadIdx.x;
  if (i >= total) return;
  int j = i & 127;
  out[i] = (acc[i] + b1[j] + b2[j] + b3[j]) * (1.0f / 3.0f);
}

// ---------- launch ----------

extern "C" void kernel_launch(void* const* d_in, const int* in_sizes, int n_in,
                              void* d_out, int out_size, void* d_ws, size_t ws_size,
                              hipStream_t stream) {
  const float* x        = (const float*)d_in[0];
  const int*   ei_par   = (const int*)d_in[1];   // [2, E] flat: src then dst
  const int*   ei_chi   = (const int*)d_in[2];
  const int*   ei_rel   = (const int*)d_in[3];
  const float* gp_W     = (const float*)d_in[4];
  const float* gp_asrc  = (const float*)d_in[5];
  const float* gp_adst  = (const float*)d_in[6];
  const float* gp_b     = (const float*)d_in[7];
  const float* gc_W     = (const float*)d_in[8];
  const float* gc_asrc  = (const float*)d_in[9];
  const float* gc_adst  = (const float*)d_in[10];
  const float* gc_b     = (const float*)d_in[11];
  const float* sg_Wn    = (const float*)d_in[12];
  const float* sg_Wr    = (const float*)d_in[13];
  const float* sg_b     = (const float*)d_in[14];
  float* out = (float*)d_out;

  // workspace carve-up (all 64B-aligned sizes)
  char* p = (char*)d_ws;
  float* acc   = (float*)p; p += (size_t)NN * DD * 4;   // 25.6 MB
  float* h     = (float*)p; p += (size_t)NN * DD * 4;   // 25.6 MB (GAT h / SAGE ssum)
  float* as_   = (float*)p; p += (size_t)NN * 4;
  float* ad_   = (float*)p; p += (size_t)NN * 4;
  int*   menc  = (int*)p;   p += (size_t)NN * 4;
  float* denom = (float*)p; p += (size_t)NN * 4;
  float* cnt   = (float*)p; p += (size_t)NN * 4;
  (void)ws_size; (void)n_in; (void)in_sizes; (void)out_size;

  const int total   = NN * DD;                 // 6,400,000
  const int edges   = EE + NN;                 // edges incl. self loops
  const int gemmGrd = (NN + 63) / 64;          // 782 blocks
  const int thrGrd  = (edges + 255) / 256;     // thread-per-edge kernels
  const int wavGrd  = (edges + 7) / 8;         // wave-per-edge (256 thr = 8 waves)
  const int relWav  = (EE + 7) / 8;
  const int elemGrd = (total + 255) / 256;
  const int nodeGrd = (NN + 255) / 256;
  const int dotGrd  = (NN * 32 + 255) / 256;   // wave per node

  const float NEG_INF = -__builtin_huge_valf();
  union { float f; int i; } ninf; ninf.f = NEG_INF;
  int enc_ninf = (ninf.i >= 0) ? ninf.i : (ninf.i ^ 0x7fffffff);

  // accumulator for (o1 + o2 + o3)
  fill_f32_kernel<<<elemGrd, 256, 0, stream>>>(acc, total, 0.0f);

  // ---- GAT #1 (is_parent) ----
  wmma_gemm_kernel<false><<<gemmGrd, 128, 0, stream>>>(x, gp_W, h, NN);
  alpha_dots_kernel<<<dotGrd, 256, 0, stream>>>(h, gp_asrc, gp_adst, as_, ad_, NN);
  fill_i32_kernel<<<nodeGrd, 256, 0, stream>>>(menc, NN, enc_ninf);
  fill_f32_kernel<<<nodeGrd, 256, 0, stream>>>(denom, NN, 0.0f);
  edge_max_kernel<<<thrGrd, 256, 0, stream>>>(ei_par, ei_par + EE, as_, ad_, menc, EE, NN);
  edge_expsum_kernel<<<thrGrd, 256, 0, stream>>>(ei_par, ei_par + EE, as_, ad_, menc, denom, EE, NN);
  edge_scatter_kernel<<<wavGrd, 256, 0, stream>>>(ei_par, ei_par + EE, as_, ad_, menc, denom, h, acc, EE, NN);

  // ---- GAT #2 (is_child) ----
  wmma_gemm_kernel<false><<<gemmGrd, 128, 0, stream>>>(x, gc_W, h, NN);
  alpha_dots_kernel<<<dotGrd, 256, 0, stream>>>(h, gc_asrc, gc_adst, as_, ad_, NN);
  fill_i32_kernel<<<nodeGrd, 256, 0, stream>>>(menc, NN, enc_ninf);
  fill_f32_kernel<<<nodeGrd, 256, 0, stream>>>(denom, NN, 0.0f);
  edge_max_kernel<<<thrGrd, 256, 0, stream>>>(ei_chi, ei_chi + EE, as_, ad_, menc, EE, NN);
  edge_expsum_kernel<<<thrGrd, 256, 0, stream>>>(ei_chi, ei_chi + EE, as_, ad_, menc, denom, EE, NN);
  edge_scatter_kernel<<<wavGrd, 256, 0, stream>>>(ei_chi, ei_chi + EE, as_, ad_, menc, denom, h, acc, EE, NN);

  // ---- SAGE (relate) ----
  wmma_gemm_kernel<true><<<gemmGrd, 128, 0, stream>>>(x, sg_Wr, acc, NN);   // acc += x @ Wr
  fill_f32_kernel<<<elemGrd, 256, 0, stream>>>(h, total, 0.0f);             // reuse h as ssum
  fill_f32_kernel<<<nodeGrd, 256, 0, stream>>>(cnt, NN, 0.0f);
  sage_sum_kernel<<<relWav, 256, 0, stream>>>(ei_rel, ei_rel + EE, x, h, cnt, EE);
  sage_mean_kernel<<<elemGrd, 256, 0, stream>>>(h, cnt, total);
  wmma_gemm_kernel<true><<<gemmGrd, 128, 0, stream>>>(h, sg_Wn, acc, NN);   // acc += mean @ Wn

  // ---- HeteroConv mean + biases ----
  finalize_kernel<<<elemGrd, 256, 0, stream>>>(acc, gp_b, gc_b, sg_b, out, total);
}
